// HybridSquidOperator_74612171866608
// MI455X (gfx1250) — compile-verified, matrix operand order checked by reading, not run on hardware
//
#include <hip/hip_runtime.h>
#include <hip/hip_bf16.h>

#define CH      6
#define MODES   16
#define BATCH   32
#define NPTS    65536
#define TSTEPS  8
#define TILE    512
#define HALO    28
#define EXT     (TILE + 2*HALO)   // 568
#define NT      (NPTS / TILE)     // 128
#define TWO_PI  6.28318530717958647692f

typedef _Float16 v16h __attribute__((ext_vector_type(16)));
typedef float    v8f  __attribute__((ext_vector_type(8)));

// Workspace layout (floats):
//   xpart : [B][NT][CH][MODES][2]  = 786432
//   X     : [B][CH][MODES][2]      = 6144
//   ZC    : [B][CH][31]            = 5952   (c0, ck[1..15], dk[1..15])
#define WS_XPART 0
#define WS_X     786432
#define WS_ZC    (786432 + 6144)

__device__ __forceinline__ float fast_tanh(float x) {
#if __has_builtin(__builtin_amdgcn_tanhf)
    return __builtin_amdgcn_tanhf(x);
#else
    float r;
    asm volatile("v_tanh_f32 %0, %1" : "=v"(r) : "v"(x));
    return r;
#endif
}

// addrspace(3) byte offset of a shared-memory pointer (AMDGPU generic LDS
// pointers carry the LDS offset in the low 32 bits).
__device__ __forceinline__ unsigned lds_offset(const void* p) {
    return (unsigned)(unsigned long long)(uintptr_t)p;
}

// ---------------------------------------------------------------------------
// Kernel A: full RK4 integration of one (batch, tile) slab in LDS, then a
// fused 16-bin partial DFT of the central 512 points. u_final never touches
// HBM. Grid: BATCH*NT blocks x 256 threads.
// ---------------------------------------------------------------------------
__global__ __launch_bounds__(256) void ode_dft_kernel(
    const float* __restrict__ u0, const float* __restrict__ ts,
    const float* __restrict__ cw, const float* __restrict__ cb,
    float* __restrict__ xpart)
{
    __shared__ float su[CH][EXT];   // current state
    __shared__ float sk[CH][EXT];   // dyn() output (k1..k4)
    __shared__ float st[CH][EXT];   // dyn() input (u + alpha*k)
    __shared__ float sa[CH][EXT];   // RK4 accumulator

    const int tid  = threadIdx.x;
    const int b    = blockIdx.x / NT;
    const int tile = blockIdx.x % NT;
    const int base = tile * TILE - HALO;    // global n of LDS x=0 (may be <0)

    // conv weights / bias -> registers (uniform)
    float w[CH][CH][3];
    float bias[CH];
    #pragma unroll
    for (int o = 0; o < CH; ++o) {
        bias[o] = cb[o];
        #pragma unroll
        for (int i = 0; i < CH; ++i)
            #pragma unroll
            for (int t = 0; t < 3; ++t)
                w[o][i][t] = cw[(o*CH + i)*3 + t];
    }

    int lo = (base < 0) ? -base : 0;
    int hi = (base + EXT > NPTS) ? (NPTS - base) : EXT;

    // Async copy of the u0 slab straight into LDS (no VGPR staging):
    // GVS mode: 64-bit SGPR base + 32-bit per-lane byte offset.
    {
        const unsigned long long sbase = (unsigned long long)(uintptr_t)u0;
        #pragma unroll
        for (int c = 0; c < CH; ++c) {
            const unsigned row = ((unsigned)(b*CH + c)) << 16;   // element row base
            for (int x = lo + tid; x < hi; x += 256) {
                const unsigned goff   = (row + (unsigned)(base + x)) * 4u;
                const unsigned ldsoff = lds_offset(&su[c][x]);
                asm volatile("global_load_async_to_lds_b32 %0, %1, %2"
                             :: "v"(ldsoff), "v"(goff), "s"(sbase)
                             : "memory");
            }
        }
        asm volatile("s_wait_asynccnt 0x0" ::: "memory");
    }
    __syncthreads();

    // dyn: out = tanh(conv1d_k3(in) + bias), valid on [l,h)
    auto dyn = [&](float (*in)[EXT], float (*out)[EXT], int l, int h) {
        for (int x = l + tid; x < h; x += 256) {
            const int n = base + x;
            float L[CH], M[CH], R[CH];
            #pragma unroll
            for (int i = 0; i < CH; ++i) {
                M[i] = in[i][x];
                L[i] = (n == 0)        ? 0.0f : in[i][x-1];
                R[i] = (n == NPTS-1)   ? 0.0f : in[i][x+1];
            }
            #pragma unroll
            for (int o = 0; o < CH; ++o) {
                float s = bias[o];
                #pragma unroll
                for (int i = 0; i < CH; ++i)
                    s = fmaf(w[o][i][0], L[i],
                        fmaf(w[o][i][1], M[i],
                        fmaf(w[o][i][2], R[i], s)));
                out[o][x] = fast_tanh(s);
            }
        }
    };

    // outb = Ab + alpha*Kb on [l,h) (aliasing outb==Ab is fine, elementwise)
    auto lincomb = [&](float (*outb)[EXT], float (*Ab)[EXT], float alpha,
                       float (*Kb)[EXT], int l, int h) {
        for (int x = l + tid; x < h; x += 256)
            #pragma unroll
            for (int c = 0; c < CH; ++c)
                outb[c][x] = fmaf(alpha, Kb[c][x], Ab[c][x]);
    };

    auto shrink = [&]() {
        if (base + lo != 0)    lo += 1;    // not at global left edge
        if (base + hi != NPTS) hi -= 1;    // not at global right edge
    };

    for (int s = 0; s < TSTEPS - 1; ++s) {
        const float dt = ts[s+1] - ts[s];

        shrink(); dyn(su, sk, lo, hi); __syncthreads();        // k1
        lincomb(sa, su, dt*(1.0f/6.0f), sk, lo, hi);           // acc = u + dt/6 k1
        lincomb(st, su, 0.5f*dt,        sk, lo, hi);           // t2
        __syncthreads();
        shrink(); dyn(st, sk, lo, hi); __syncthreads();        // k2
        lincomb(sa, sa, dt*(1.0f/3.0f), sk, lo, hi);
        lincomb(st, su, 0.5f*dt,        sk, lo, hi);           // t3
        __syncthreads();
        shrink(); dyn(st, sk, lo, hi); __syncthreads();        // k3
        lincomb(sa, sa, dt*(1.0f/3.0f), sk, lo, hi);
        lincomb(st, su, dt,             sk, lo, hi);           // t4
        __syncthreads();
        shrink(); dyn(st, sk, lo, hi); __syncthreads();        // k4
        lincomb(su, sa, dt*(1.0f/6.0f), sk, lo, hi);           // u <- next state
        __syncthreads();
    }

    // fused partial DFT: 16 bins over the central 512 points ([lo,hi) is
    // exactly the central tile for every block). Thread (c,k) pairs.
    if (tid < CH*MODES) {
        const int c = tid / MODES;
        const int k = tid % MODES;
        const int n0 = base + lo;
        // exact integer phase reduction: phase0 = 2*pi*((k*n0) mod N)/N
        const int r0 = (int)(((long long)k * (long long)n0) % NPTS);
        const float phi0 = TWO_PI * (float)r0 / (float)NPTS;
        const float dphi = TWO_PI * (float)k  / (float)NPTS;
        float cc, ss, cd, sd;
        __sincosf(phi0, &ss, &cc);
        __sincosf(dphi, &sd, &cd);
        float re = 0.0f, im = 0.0f;
        for (int x = lo; x < hi; ++x) {
            const float u = su[c][x];
            re = fmaf(u,  cc, re);
            im = fmaf(u, -ss, im);
            const float nc = cc*cd - ss*sd;     // rotate e^{-i*phi} basis
            ss = fmaf(ss, cd, cc*sd);
            cc = nc;
        }
        const size_t idx = ((((size_t)b*NT + tile)*CH + c)*MODES + k) * 2;
        xpart[idx]     = re;
        xpart[idx + 1] = im;
    }
}

// ---------------------------------------------------------------------------
// Kernel B: deterministic reduction of per-tile DFT partials.
// One thread per (b,c,k).
// ---------------------------------------------------------------------------
__global__ void reduce_kernel(const float* __restrict__ xpart,
                              float* __restrict__ X)
{
    const int t = blockIdx.x*blockDim.x + threadIdx.x;
    if (t >= BATCH*CH*MODES) return;
    const int b = t / (CH*MODES);
    const int r = t % (CH*MODES);
    float re = 0.0f, im = 0.0f;
    for (int tile = 0; tile < NT; ++tile) {
        const size_t idx = (((size_t)b*NT + tile)*(CH*MODES) + r) * 2;
        re += xpart[idx];
        im += xpart[idx + 1];
    }
    X[(size_t)t*2]     = re;
    X[(size_t)t*2 + 1] = im;
}

// ---------------------------------------------------------------------------
// Kernel C: complex mode mixing (Y = X * W) then channel projection
// Z = proj_w x Y done with V_WMMA_F32_16X16X32_F16 (M=o pad16, K=c pad32,
// N=32 (mode,re/im) columns in two 16-wide tiles). Scales 2/N, -2/N, 1/N and
// proj_b are folded into the stored synthesis coefficients ZC.
// Grid: BATCH blocks x 128 threads (wave 0 runs the WMMA with full EXEC).
// ---------------------------------------------------------------------------
__global__ __launch_bounds__(128) void mix_project_kernel(
    const float* __restrict__ X,  const float* __restrict__ swr,
    const float* __restrict__ swi, const float* __restrict__ pw,
    const float* __restrict__ pb,  float* __restrict__ ZC)
{
    __shared__ float Y[CH][2*MODES];   // [c][2k+part]
    __shared__ float P[16][8];         // padded proj_w
    const int b = blockIdx.x;
    const int t = threadIdx.x;

    {   // padded projection matrix
        const int row = t % 16, col = t / 16;          // 128 threads -> 16x8
        P[row][col] = (row < CH && col < CH) ? pw[row*CH + col] : 0.0f;
    }
    if (t < CH*MODES) {                                 // Y[c][k] complex
        const int c = t / MODES, k = t % MODES;
        float yr = 0.0f, yi = 0.0f;
        #pragma unroll
        for (int i = 0; i < CH; ++i) {
            const float xr = X[((b*CH + i)*MODES + k)*2];
            const float xi = X[((b*CH + i)*MODES + k)*2 + 1];
            const float wr = swr[(i*CH + c)*MODES + k];
            const float wi = swi[(i*CH + c)*MODES + k];
            yr = fmaf(xr, wr, fmaf(-xi, wi, yr));
            yi = fmaf(xr, wi, fmaf( xi, wr, yi));
        }
        Y[c][2*k]     = yr;
        Y[c][2*k + 1] = yi;
    }
    __syncthreads();

    if (t < 32) {                       // wave 0, EXEC all ones
        const int  lane = t;
        const int  row  = lane & 15;
        const bool lowh = lane < 16;

        // A: 16x32 f16, documented layout: lanes 0-15 hold K=0..7,16..23,
        // lanes 16-31 hold K=8..15,24..31. Only K<6 (lanes 0-15) nonzero.
        // Branchless: unconditional clamped LDS reads + arithmetic select.
        v16h a;
        #pragma unroll
        for (int i = 0; i < 16; ++i) {
            const float pv = P[row][(i < 8) ? i : 0];
            a[i] = (_Float16)((lowh && i < CH) ? pv : 0.0f);
        }

        #pragma unroll
        for (int ntile = 0; ntile < 2; ++ntile) {
            const int col = row + 16*ntile;            // output column j
            v16h bb;                                   // B: K striped like A
            #pragma unroll
            for (int i = 0; i < 16; ++i) {
                const float yv = Y[(i < CH) ? i : 0][col];
                bb[i] = (_Float16)((lowh && i < CH) ? yv : 0.0f);
            }
            v8f acc = {};
            acc = __builtin_amdgcn_wmma_f32_16x16x32_f16(
                      false, a, false, bb, (short)0, acc, false, false);

            // D layout: lane = column, VGPR r = row M=r (lanes 0-15 rows 0-7)
            if (lowh) {
                const int j = lane + 16*ntile;
                const int k = j >> 1, part = j & 1;
                #pragma unroll
                for (int o = 0; o < CH; ++o) {
                    const float  d    = acc[o];
                    const size_t zidx = ((size_t)b*CH + o)*31;
                    if (k == 0) {
                        if (part == 0)
                            ZC[zidx] = d * (1.0f/NPTS) + pb[o];  // DC (imag dropped)
                    } else if (part == 0) {
                        ZC[zidx + k]      =  d * (2.0f/NPTS);    // cos coeff
                    } else {
                        ZC[zidx + 15 + k] = -d * (2.0f/NPTS);    // sin coeff
                    }
                }
            }
        }
    }
}

// ---------------------------------------------------------------------------
// Kernel D: 16-mode Fourier synthesis, coalesced 50MB store.
// Grid: (NPTS/256, BATCH) x 256.
// ---------------------------------------------------------------------------
__global__ __launch_bounds__(256) void synth_kernel(
    const float* __restrict__ ZC, float* __restrict__ out)
{
    const int n = blockIdx.x*256 + threadIdx.x;
    const int b = blockIdx.y;
    const float* z = ZC + (size_t)b*CH*31;

    const float th = TWO_PI * (float)n / (float)NPTS;
    float c1, s1;
    __sincosf(th, &s1, &c1);

    float acc[CH];
    #pragma unroll
    for (int o = 0; o < CH; ++o) acc[o] = z[o*31];

    float ck = c1, sk = s1;
    #pragma unroll
    for (int k = 1; k <= 15; ++k) {
        #pragma unroll
        for (int o = 0; o < CH; ++o)
            acc[o] = fmaf(z[o*31 + k], ck, fmaf(z[o*31 + 15 + k], sk, acc[o]));
        const float nc = ck*c1 - sk*s1;     // chained rotation -> next harmonic
        sk = fmaf(sk, c1, ck*s1);
        ck = nc;
    }
    #pragma unroll
    for (int o = 0; o < CH; ++o)
        out[(((size_t)b*CH + o) << 16) + n] = acc[o];
}

// ---------------------------------------------------------------------------
extern "C" void kernel_launch(void* const* d_in, const int* in_sizes, int n_in,
                              void* d_out, int out_size, void* d_ws, size_t ws_size,
                              hipStream_t stream)
{
    (void)in_sizes; (void)n_in; (void)out_size; (void)ws_size;
    const float* u0  = (const float*)d_in[0];
    const float* ts  = (const float*)d_in[1];
    const float* cw  = (const float*)d_in[2];
    const float* cb  = (const float*)d_in[3];
    const float* swr = (const float*)d_in[4];
    const float* swi = (const float*)d_in[5];
    const float* pw  = (const float*)d_in[6];
    const float* pb  = (const float*)d_in[7];
    float* out = (float*)d_out;
    float* ws  = (float*)d_ws;

    float* xpart = ws + WS_XPART;
    float* X     = ws + WS_X;
    float* ZC    = ws + WS_ZC;

    ode_dft_kernel<<<BATCH*NT, 256, 0, stream>>>(u0, ts, cw, cb, xpart);
    reduce_kernel<<<(BATCH*CH*MODES + 255)/256, 256, 0, stream>>>(xpart, X);
    mix_project_kernel<<<BATCH, 128, 0, stream>>>(X, swr, swi, pw, pb, ZC);
    synth_kernel<<<dim3(NPTS/256, BATCH), 256, 0, stream>>>(ZC, out);
}
